// JointNet_25907242730051
// MI455X (gfx1250) — matrix operand order repaired and checked by reading, not compiled, not run on hardware
//
#include <hip/hip_runtime.h>
#include <math.h>

#define BB 4
#define TT 320
#define UU 80
#define DD 512
#define INNER 512
#define VOCAB 6
#define MAXRLE 50
#define NC 56          // real output channels
#define NCP 64         // padded to 4 WMMA N-tiles
#define WC_STRIDE 520  // halves per padded Wc row in LDS (bank-conflict-free)

// Use CDNA5 async global->LDS DMA for the weight staging. If the assembler
// rejects the mnemonic, set to 0 to fall back to plain loads+ds_stores.
#define USE_ASYNC_LDS 1

typedef __attribute__((ext_vector_type(16))) _Float16 v16h;
typedef __attribute__((ext_vector_type(8)))  _Float16 v8h;
typedef __attribute__((ext_vector_type(2)))  __fp16   fp16x2;  // cvt_pkrtz return type
typedef __attribute__((ext_vector_type(8)))  float    v8f;

// Branch-free tanh: 1 - 2/(exp(2x)+1). v_exp_f32 + v_rcp_f32, saturates at +-1.
__device__ __forceinline__ float fast_tanh(float x) {
    float e = __builtin_amdgcn_exp2f(x * 2.88539008177793f);  // exp(2x)
    return 1.0f - 2.0f * __builtin_amdgcn_rcpf(e + 1.0f);
}

// ---------------------------------------------------------------------------
// Projection: P[n][i] = dot(W_f[i][koff:koff+512], X[n][:]) (+ bias[i])
// One block per row n; 256 threads x 2 outputs each = 512 outputs.
// ---------------------------------------------------------------------------
__global__ void joint_proj_kernel(const float* __restrict__ X,
                                  const float* __restrict__ Wf,
                                  const float* __restrict__ bias,
                                  float* __restrict__ P,
                                  int koff, int useBias) {
    __shared__ float xs[DD];
    const int n = blockIdx.x;
    const float* xrow = X + (size_t)n * DD;
    for (int k = threadIdx.x; k < DD; k += 256) xs[k] = xrow[k];
    __syncthreads();

    const int i0 = threadIdx.x * 2;
    const float* w0 = Wf + (size_t)i0 * (2 * DD) + koff;
    const float* w1 = w0 + 2 * DD;
    float acc0 = 0.f, acc1 = 0.f;
#pragma unroll 4
    for (int k = 0; k < DD; k += 4) {
        float4 a = *(const float4*)(w0 + k);
        float4 b = *(const float4*)(w1 + k);
        float4 x = *(const float4*)(xs + k);
        acc0 += a.x * x.x + a.y * x.y + a.z * x.z + a.w * x.w;
        acc1 += b.x * x.x + b.y * x.y + b.z * x.z + b.w * x.w;
    }
    if (useBias) { acc0 += bias[i0]; acc1 += bias[i0 + 1]; }
    P[(size_t)n * INNER + i0]     = acc0;
    P[(size_t)n * INNER + i0 + 1] = acc1;
}

// ---------------------------------------------------------------------------
// Pack [W_base; W_rle] -> f16 Wc[64][512] (rows 56..63 zero), bias bc[64].
// ---------------------------------------------------------------------------
__global__ void joint_prep_wc_kernel(const float* __restrict__ Wb,
                                     const float* __restrict__ bb,
                                     const float* __restrict__ Wr,
                                     const float* __restrict__ br,
                                     _Float16* __restrict__ Wc,
                                     float* __restrict__ bc) {
    const int gid = blockIdx.x * 256 + threadIdx.x;
    if (gid < NCP * INNER) {
        const int row = gid >> 9;
        const int col = gid & (INNER - 1);
        float v = 0.f;
        if (row < VOCAB)      v = Wb[row * INNER + col];
        else if (row < NC)    v = Wr[(row - VOCAB) * INNER + col];
        Wc[gid] = (_Float16)v;
    }
    if (gid < NCP) {
        float v = 0.f;
        if (gid < VOCAB)      v = bb[gid];
        else if (gid < NC)    v = br[gid - VOCAB];
        bc[gid] = v;
    }
}

// ---------------------------------------------------------------------------
// Fused: h = tanh(P_enc[b,t] + P_dec[b,u]);  out = [W_base;W_rle] @ h + bias
// Block = 256 threads (8 waves). Wave w: t = 8*blockIdx.y + w, 16 u's.
// A fragment (16x32 f16) built in-register per documented CDNA5 layout;
// B fragments read from LDS-staged Wc; 4x v_wmma_f32_16x16x32_f16 per K-chunk.
// ---------------------------------------------------------------------------
__global__ void joint_fused_kernel(const float* __restrict__ Penc,
                                   const float* __restrict__ Pdec,
                                   const _Float16* __restrict__ Wc,
                                   const float* __restrict__ bc,
                                   float* __restrict__ out_base,
                                   float* __restrict__ out_rle) {
    extern __shared__ char smem[];
    _Float16* swc = (_Float16*)smem;

    // ---- Stage Wc (64x512 f16 = 64 KB) into LDS with padded row stride.
#if USE_ASYNC_LDS
    {
        // Dynamic LDS starts at offset 0 (no static LDS in this kernel), so
        // the LDS byte offset can be passed directly in the VDST VGPR.
        // 4096 chunks of 16 B; 16 per thread. ASYNCcnt-tracked DMA, no VGPR data.
        const char* gsrc = (const char*)Wc;
        for (int idx = threadIdx.x; idx < NCP * 64; idx += 256) {
            const int row = idx >> 6;                    // 64 x 16B per row
            const int col = idx & 63;
            unsigned int  ldsoff = (unsigned int)(row * (WC_STRIDE * 2) + col * 16);
            unsigned long long gaddr = (unsigned long long)(gsrc + (size_t)idx * 16);
            asm volatile("global_load_async_to_lds_b128 %0, %1, off"
                         :: "v"(ldsoff), "v"(gaddr)
                         : "memory");
        }
        asm volatile("s_wait_asynccnt 0" ::: "memory");
    }
#else
    {
        const uint32_t* src = (const uint32_t*)Wc;
        uint32_t* dst = (uint32_t*)swc;
        for (int idx = threadIdx.x; idx < NCP * (INNER / 2); idx += 256) {
            const int row = idx >> 8;     // 256 dwords per source row
            const int col = idx & 255;
            dst[row * (WC_STRIDE / 2) + col] = src[idx];
        }
    }
#endif
    __syncthreads();

    const int b    = blockIdx.z;
    const int t    = blockIdx.y * 8 + (threadIdx.x >> 5);
    const int u0   = blockIdx.x * 16;
    const int lane = threadIdx.x & 31;
    const int laneM  = lane & 15;
    const int laneHi = lane >> 4;

    const float* pe = Penc + (size_t)(b * TT + t) * INNER;
    const float* pd = Pdec + (size_t)(b * UU + u0 + laneM) * INNER;

    v8f c0 = {}, c1 = {}, c2 = {}, c3 = {};

    for (int kc = 0; kc < INNER; kc += 32) {
        // hint next chunk of the streaming operand
        __builtin_prefetch(pd + kc + 32, 0, 0);

        // ----- A fragment: lane holds row M=laneM; two contiguous K-runs of 8
        const int kb0 = kc + laneHi * 8;        // halves 0..7  : K = kb0..kb0+7
        const int kb1 = kc + 16 + laneHi * 8;   // halves 8..15 : K = kb1..kb1+7
        float4 e0 = *(const float4*)(pe + kb0);
        float4 e1 = *(const float4*)(pe + kb0 + 4);
        float4 e2 = *(const float4*)(pe + kb1);
        float4 e3 = *(const float4*)(pe + kb1 + 4);
        float4 d0 = *(const float4*)(pd + kb0);
        float4 d1 = *(const float4*)(pd + kb0 + 4);
        float4 d2 = *(const float4*)(pd + kb1);
        float4 d3 = *(const float4*)(pd + kb1 + 4);

        union { v16h v; fp16x2 p[8]; } au;
        au.p[0] = __builtin_amdgcn_cvt_pkrtz(fast_tanh(e0.x + d0.x), fast_tanh(e0.y + d0.y));
        au.p[1] = __builtin_amdgcn_cvt_pkrtz(fast_tanh(e0.z + d0.z), fast_tanh(e0.w + d0.w));
        au.p[2] = __builtin_amdgcn_cvt_pkrtz(fast_tanh(e1.x + d1.x), fast_tanh(e1.y + d1.y));
        au.p[3] = __builtin_amdgcn_cvt_pkrtz(fast_tanh(e1.z + d1.z), fast_tanh(e1.w + d1.w));
        au.p[4] = __builtin_amdgcn_cvt_pkrtz(fast_tanh(e2.x + d2.x), fast_tanh(e2.y + d2.y));
        au.p[5] = __builtin_amdgcn_cvt_pkrtz(fast_tanh(e2.z + d2.z), fast_tanh(e2.w + d2.w));
        au.p[6] = __builtin_amdgcn_cvt_pkrtz(fast_tanh(e3.x + d3.x), fast_tanh(e3.y + d3.y));
        au.p[7] = __builtin_amdgcn_cvt_pkrtz(fast_tanh(e3.z + d3.z), fast_tanh(e3.w + d3.w));
        v16h a = au.v;

        // ----- B fragments: lane holds col N=laneM; K = kc + laneHi*8 + j (j<8), kc + 16 + laneHi*8 + (j-8) (j>=8)
        const int kB = kc + laneHi * 8;
        const _Float16* w0p = swc + (size_t)(0 * 16 + laneM) * WC_STRIDE + kB;
        const _Float16* w1p = swc + (size_t)(1 * 16 + laneM) * WC_STRIDE + kB;
        const _Float16* w2p = swc + (size_t)(2 * 16 + laneM) * WC_STRIDE + kB;
        const _Float16* w3p = swc + (size_t)(3 * 16 + laneM) * WC_STRIDE + kB;
        union { v16h v; v8h h[2]; } b0u, b1u, b2u, b3u;
        b0u.h[0] = *(const v8h*)w0p;  b0u.h[1] = *(const v8h*)(w0p + 16);
        b1u.h[0] = *(const v8h*)w1p;  b1u.h[1] = *(const v8h*)(w1p + 16);
        b2u.h[0] = *(const v8h*)w2p;  b2u.h[1] = *(const v8h*)(w2p + 16);
        b3u.h[0] = *(const v8h*)w3p;  b3u.h[1] = *(const v8h*)(w3p + 16);

        c0 = __builtin_amdgcn_wmma_f32_16x16x32_f16(false, a, false, b0u.v, (short)0, c0, false, false);
        c1 = __builtin_amdgcn_wmma_f32_16x16x32_f16(false, a, false, b1u.v, (short)0, c1, false, false);
        c2 = __builtin_amdgcn_wmma_f32_16x16x32_f16(false, a, false, b2u.v, (short)0, c2, false, false);
        c3 = __builtin_amdgcn_wmma_f32_16x16x32_f16(false, a, false, b3u.v, (short)0, c3, false, false);
    }

    // ----- Epilogue: D layout is VGPR r -> M = laneHi*8 + r, col N = laneM.
    const float bc0 = bc[ 0 + laneM];
    const float bc1 = bc[16 + laneM];
    const float bc2 = bc[32 + laneM];
    const float bc3 = bc[48 + laneM];
    const size_t posbase = ((size_t)(b * TT + t)) * UU + u0;

    v8f cc[4] = {c0, c1, c2, c3};
    const float bcv[4] = {bc0, bc1, bc2, bc3};
#pragma unroll
    for (int n = 0; n < 4; ++n) {
        const int N = n * 16 + laneM;
#pragma unroll
        for (int r = 0; r < 8; ++r) {
            const int M = laneHi * 8 + r;
            const size_t pos = posbase + M;        // u = u0 + M
            const float v = cc[n][r] + bcv[n];
            if (N < VOCAB) {
                out_base[pos * VOCAB + N] = v;
            } else if (N < NC) {
                out_rle[pos * MAXRLE + (N - VOCAB)] = v;
            }
        }
    }
}

// ---------------------------------------------------------------------------
extern "C" void kernel_launch(void* const* d_in, const int* in_sizes, int n_in,
                              void* d_out, int out_size, void* d_ws, size_t ws_size,
                              hipStream_t stream) {
    const float* enc    = (const float*)d_in[0];   // [B,T,D]
    const float* dec    = (const float*)d_in[1];   // [B,U,D]
    const float* W_f    = (const float*)d_in[2];   // [INNER, 2D]
    const float* b_f    = (const float*)d_in[3];   // [INNER]
    const float* W_base = (const float*)d_in[4];   // [VOCAB, INNER]
    const float* b_base = (const float*)d_in[5];   // [VOCAB]
    const float* W_rle  = (const float*)d_in[6];   // [MAXRLE, INNER]
    const float* b_rle  = (const float*)d_in[7];   // [MAXRLE]

    char* ws = (char*)d_ws;
    float*    Penc = (float*)ws;                                    // 1280*512 f32
    float*    Pdec = (float*)(ws + (size_t)BB * TT * INNER * 4);    // 320*512 f32
    _Float16* Wc   = (_Float16*)(ws + (size_t)BB * TT * INNER * 4
                                    + (size_t)BB * UU * INNER * 4); // 64*512 f16
    float*    bc   = (float*)((char*)Wc + (size_t)NCP * INNER * 2); // 64 f32

    float* out_base = (float*)d_out;
    float* out_rle  = out_base + (size_t)BB * TT * UU * VOCAB;

    // 1) factored projections of the joint pre-activation
    joint_proj_kernel<<<BB * TT, 256, 0, stream>>>(enc, W_f, b_f, Penc, 0, 0);
    joint_proj_kernel<<<BB * UU, 256, 0, stream>>>(dec, W_f, b_f, Pdec, DD, 1);

    // 2) pack output weights/bias (f16, padded to 64 channels)
    joint_prep_wc_kernel<<<(NCP * INNER + 255) / 256, 256, 0, stream>>>(
        W_base, b_base, W_rle, b_rle, Wc, bc);

    // 3) fused tanh + WMMA output GEMM
    dim3 grid(UU / 16, TT / 8, BB);
    joint_fused_kernel<<<grid, 256, NCP * WC_STRIDE * 2, stream>>>(
        Penc, Pdec, Wc, bc, out_base, out_rle);
}
